// DALLETransformer_16758962389476
// MI455X (gfx1250) — matrix operand order repaired and checked by reading, not compiled, not run on hardware
//
#include <hip/hip_runtime.h>
#include <hip/hip_bf16.h>

typedef __attribute__((ext_vector_type(16))) _Float16 v16h;
typedef __attribute__((ext_vector_type(8)))  float    v8f;

#define BB   8
#define TT   64
#define IIM  1024
#define SS   1088
#define DD   1024
#define HH   16
#define HDD  64
#define LLAY 12
#define FFF  4096
#define VTT  16384
#define VII  8192

enum { ACT_NONE = 0, ACT_GELU = 1 };

__device__ __forceinline__ v8f wmma32(v16h a, v16h b, v8f c) {
  // (neg_a, A, neg_b, B, c_mod, C, reuse_a, reuse_b)
  return __builtin_amdgcn_wmma_f32_16x16x32_f16(false, a, false, b, (short)0, c, false, false);
}

// K-offset pattern for 16-bit A fragment (contiguous order, matches B fills):
// lane-half hi, VGPR pair p holds K = {2p, 2p+1} + 16*hi
__device__ __forceinline__ int akoff(int p, int hi) {
  return 2 * p + 16 * hi;
}

// --------- CDNA5 async memory->LDS copy (ASYNCcnt-tracked, ISA 08 §4) ------
// LDS offset = low 32 bits of the generic address of a __shared__ object
// (ISA 10.2: LDS aperture addresses truncate to addr[31:0]).
__device__ __forceinline__ unsigned lds_off_of(const void* p) {
  return (unsigned)(uintptr_t)p;
}
__device__ __forceinline__ void async_copy_b128(unsigned lds_byte_off, const void* gaddr) {
  asm volatile("global_load_async_to_lds_b128 %0, %1, off"
               :
               : "v"(lds_byte_off), "v"(gaddr)
               : "memory");
}
__device__ __forceinline__ void async_wait0() {
  asm volatile("s_wait_asynccnt 0x0" ::: "memory");
}

// ---------------------------------------------------------------- utilities
__global__ void cvt_f32_to_f16_kernel(const float* __restrict__ in,
                                      _Float16* __restrict__ out, size_t n) {
  size_t i = (size_t)blockIdx.x * blockDim.x + threadIdx.x;
  if (i < n) out[i] = (_Float16)in[i];
}

__global__ void embed_kernel(const int* __restrict__ text, const int* __restrict__ img,
                             const float* __restrict__ tok_emb, const float* __restrict__ pos_emb,
                             float* __restrict__ x) {
  size_t idx = (size_t)blockIdx.x * blockDim.x + threadIdx.x;
  if (idx >= (size_t)BB * SS * DD) return;
  int d = (int)(idx % DD);
  int s = (int)((idx / DD) % SS);
  int b = (int)(idx / ((size_t)DD * SS));
  int tok = (s < TT) ? text[b * TT + s] : (img[b * IIM + (s - TT)] + VTT);
  x[idx] = tok_emb[(size_t)tok * DD + d] + pos_emb[(size_t)s * DD + d];
}

// LayerNorm over D=1024, one token per 256-thread block, writes f16
__global__ __launch_bounds__(256) void ln_kernel(const float* __restrict__ x,
                                                 const float* __restrict__ g,
                                                 const float* __restrict__ bta,
                                                 _Float16* __restrict__ out) {
  __shared__ float red[256];
  const int row = blockIdx.x;
  const int tid = threadIdx.x;
  const float* xr = x + (size_t)row * DD;

  float s = 0.f;
  for (int i = tid; i < DD; i += 256) s += xr[i];
  red[tid] = s; __syncthreads();
  for (int o = 128; o > 0; o >>= 1) { if (tid < o) red[tid] += red[tid + o]; __syncthreads(); }
  float mu = red[0] * (1.0f / DD);
  __syncthreads();

  float v = 0.f;
  for (int i = tid; i < DD; i += 256) { float d = xr[i] - mu; v += d * d; }
  red[tid] = v; __syncthreads();
  for (int o = 128; o > 0; o >>= 1) { if (tid < o) red[tid] += red[tid + o]; __syncthreads(); }
  float rs = rsqrtf(red[0] * (1.0f / DD) + 1e-5f);

  for (int i = tid; i < DD; i += 256)
    out[(size_t)row * DD + i] = (_Float16)((xr[i] - mu) * rs * g[i] + bta[i]);
}

// Split qkv16 [B,S,3,H,HD] into q/k/v [B,H,S,HD]
__global__ void repack_qkv_kernel(const _Float16* __restrict__ qkv,
                                  _Float16* __restrict__ q, _Float16* __restrict__ k,
                                  _Float16* __restrict__ v) {
  size_t idx = (size_t)blockIdx.x * blockDim.x + threadIdx.x;
  if (idx >= (size_t)BB * SS * DD) return;
  int d = (int)(idx % DD);
  int s = (int)((idx / DD) % SS);
  int b = (int)(idx / ((size_t)DD * SS));
  int h = d / HDD, hd = d % HDD;
  size_t dst = (((size_t)(b * HH + h)) * SS + s) * HDD + hd;
  size_t base = ((size_t)(b * SS + s) * 3) * DD + d;
  q[dst] = qkv[base];
  k[dst] = qkv[base + DD];
  v[dst] = qkv[base + 2 * DD];
}

// Gather final-LN rows for the output positions (s = T-1 .. S-2)
__global__ void gather_head_rows_kernel(const _Float16* __restrict__ h16,
                                        _Float16* __restrict__ hsel) {
  size_t idx = (size_t)blockIdx.x * blockDim.x + threadIdx.x;
  if (idx >= (size_t)BB * IIM * DD) return;
  int d = (int)(idx % DD);
  int j = (int)((idx / DD) % IIM);
  int b = (int)(idx / ((size_t)DD * IIM));
  hsel[idx] = h16[((size_t)(b * SS + (TT - 1) + j)) * DD + d];
}

// ---------------------------------------------------------------- GEMM
// C[M,N] = A[M,K](f16) * B[K,N](f16) [+bias] [gelu] [+res(f32)] -> f32/f16
// Block: 256 thr (8 waves) computes 128x128. Wave: 32x64 (2x4 WMMA tiles).
// LDS double-buffered, filled with global_load_async_to_lds_b128 (ASYNCcnt).
template <int ACT, bool HAS_BIAS, bool HAS_RES, bool OUT32, bool OUT16>
__global__ __launch_bounds__(256) void gemm_kernel(
    const _Float16* __restrict__ A, const _Float16* __restrict__ Bm,
    const float* __restrict__ bias, const float* __restrict__ res,
    float* __restrict__ C32, _Float16* __restrict__ C16, int M, int N, int K) {
  __shared__ _Float16 As[2][128 * 40];   // 128 rows x 32 halves (+8 pad)
  __shared__ _Float16 Bs[2][32 * 136];   // 32 rows  x 128 halves (+8 pad)
  const int tid = threadIdx.x;
  const int wave = tid >> 5, lane = tid & 31;
  const int hi = lane >> 4, l16 = lane & 15;
  const int bm = blockIdx.y * 128, bn = blockIdx.x * 128;
  const int wm = (wave & 3) * 32, wn = (wave >> 2) * 64;

  v8f acc[2][4];
#pragma unroll
  for (int i = 0; i < 2; ++i)
#pragma unroll
    for (int j = 0; j < 4; ++j)
#pragma unroll
      for (int r = 0; r < 8; ++r) acc[i][j][r] = 0.f;

  const int arow = tid >> 1, apart = (tid & 1) * 16;  // 16 halves each
  const int brow = tid >> 3, bcol = (tid & 7) * 16;   // 16 halves each

  const _Float16* agp = A + (size_t)(bm + arow) * K + apart;
  const _Float16* bgp = Bm + (size_t)brow * N + bn + bcol;

  auto stage = [&](int k0, int buf) {
    const _Float16* ag = agp + k0;
    unsigned al = lds_off_of(&As[buf][arow * 40 + apart]);
    async_copy_b128(al, ag);
    async_copy_b128(al + 16, ag + 8);
    const _Float16* bg = bgp + (size_t)k0 * N;
    unsigned bl = lds_off_of(&Bs[buf][brow * 136 + bcol]);
    async_copy_b128(bl, bg);
    async_copy_b128(bl + 16, bg + 8);
  };

  stage(0, 0);
  async_wait0();
  __syncthreads();

  int buf = 0;
  for (int k0 = 0; k0 < K; k0 += 32) {
    if (k0 + 32 < K) stage(k0 + 32, buf ^ 1);

    v16h a[2], b[4];
#pragma unroll
    for (int i = 0; i < 2; ++i) {
      const _Float16* ar = &As[buf][(wm + i * 16 + l16) * 40];
#pragma unroll
      for (int p = 0; p < 8; ++p) {
        int k = akoff(p, hi);
        a[i][2 * p] = ar[k];
        a[i][2 * p + 1] = ar[k + 1];
      }
    }
#pragma unroll
    for (int j = 0; j < 4; ++j) {
      const _Float16* br = &Bs[buf][16 * hi * 136 + wn + j * 16 + l16];
#pragma unroll
      for (int e = 0; e < 16; ++e) b[j][e] = br[(size_t)e * 136];
    }
#pragma unroll
    for (int i = 0; i < 2; ++i)
#pragma unroll
      for (int j = 0; j < 4; ++j) acc[i][j] = wmma32(a[i], b[j], acc[i][j]);

    async_wait0();
    __syncthreads();
    buf ^= 1;
  }

#pragma unroll
  for (int i = 0; i < 2; ++i) {
#pragma unroll
    for (int j = 0; j < 4; ++j) {
      int gn = bn + wn + j * 16 + l16;
#pragma unroll
      for (int r = 0; r < 8; ++r) {
        int gm = bm + wm + i * 16 + r + 8 * hi;
        float v = acc[i][j][r];
        if (HAS_BIAS) v += bias[gn];
        if (ACT == ACT_GELU) v = 0.5f * v * (1.0f + erff(v * 0.70710678118f));
        size_t idx = (size_t)gm * N + gn;
        if (HAS_RES) v += res[idx];
        if (OUT32) C32[idx] = v;
        if (OUT16) C16[idx] = (_Float16)v;
      }
    }
  }
}

// ---------------------------------------------------------------- attention
// One wave per (b, h, 16 q-rows). Scores staged in dynamic LDS (16 x 1088 f32).
__global__ __launch_bounds__(32) void attn_kernel(const _Float16* __restrict__ q16,
                                                  const _Float16* __restrict__ k16,
                                                  const _Float16* __restrict__ v16,
                                                  _Float16* __restrict__ o16) {
  extern __shared__ float sc[];  // [16][SS]
  const int nqt = SS / 16;
  const int qt = blockIdx.x % nqt;
  const int h = (blockIdx.x / nqt) % HH;
  const int b = blockIdx.x / (nqt * HH);
  const int q0 = qt * 16;
  const int lane = threadIdx.x;
  const int hi = lane >> 4, l16 = lane & 15;
  const float scale = 0.125f;  // 1/sqrt(64)

  const _Float16* qb = q16 + (size_t)(b * HH + h) * SS * HDD;
  const _Float16* kb = k16 + (size_t)(b * HH + h) * SS * HDD;
  const _Float16* vb = v16 + (size_t)(b * HH + h) * SS * HDD;

  v16h a0, a1;
  {
    const _Float16* qr = qb + (size_t)(q0 + l16) * HDD;
#pragma unroll
    for (int p = 0; p < 8; ++p) {
      int k = akoff(p, hi);
      a0[2 * p] = qr[k];       a0[2 * p + 1] = qr[k + 1];
      a1[2 * p] = qr[32 + k];  a1[2 * p + 1] = qr[32 + k + 1];
    }
  }

  // scores = Q K^T * scale, causal-masked, into LDS
  for (int n0 = 0; n0 <= q0; n0 += 16) {
    const _Float16* kr = kb + (size_t)(n0 + l16) * HDD + 16 * hi;
    v16h b0, b1;
#pragma unroll
    for (int e = 0; e < 16; ++e) { b0[e] = kr[e]; b1[e] = kr[32 + e]; }
    v8f s;
#pragma unroll
    for (int r = 0; r < 8; ++r) s[r] = 0.f;
    s = wmma32(a0, b0, s);
    s = wmma32(a1, b1, s);
    int col = n0 + l16;
#pragma unroll
    for (int r = 0; r < 8; ++r) {
      int lrow = r + 8 * hi;
      float v = s[r] * scale;
      if (col > q0 + lrow) v = -1e30f;
      sc[lrow * SS + col] = v;
    }
  }
  __syncthreads();

  const int jmax = (q0 + 16 + 31) & ~31;  // round K-extent up to 32
  if (lane < 16) {
    float* sr = sc + lane * SS;
    int len = q0 + lane + 1;
    float m = -1e30f;
    for (int j = 0; j < len; ++j) m = fmaxf(m, sr[j]);
    float sum = 0.f;
    for (int j = 0; j < len; ++j) { float e = __expf(sr[j] - m); sr[j] = e; sum += e; }
    float inv = 1.0f / sum;
    for (int j = 0; j < len; ++j) sr[j] *= inv;
    for (int j = len; j < jmax; ++j) sr[j] = 0.f;
  }
  __syncthreads();

  // O = P V  (HD=64 -> 4 N-tiles of 16)
  v8f o[4];
#pragma unroll
  for (int t = 0; t < 4; ++t)
#pragma unroll
    for (int r = 0; r < 8; ++r) o[t][r] = 0.f;

  for (int kt = 0; kt < jmax; kt += 32) {
    v16h pa;
    const float* pr = sc + l16 * SS + kt;
#pragma unroll
    for (int p = 0; p < 8; ++p) {
      int k = akoff(p, hi);
      pa[2 * p] = (_Float16)pr[k];
      pa[2 * p + 1] = (_Float16)pr[k + 1];
    }
#pragma unroll
    for (int t = 0; t < 4; ++t) {
      const _Float16* vr = vb + (size_t)(kt + 16 * hi) * HDD + t * 16 + l16;
      v16h bv;
#pragma unroll
      for (int e = 0; e < 16; ++e) bv[e] = vr[(size_t)e * HDD];
      o[t] = wmma32(pa, bv, o[t]);
    }
  }

#pragma unroll
  for (int t = 0; t < 4; ++t)
#pragma unroll
    for (int r = 0; r < 8; ++r) {
      int grow = q0 + r + 8 * hi;
      o16[((size_t)(b * SS + grow) * HH + h) * HDD + t * 16 + l16] = (_Float16)o[t][r];
    }
}

// ---------------------------------------------------------------- launch
static inline unsigned blocks_for(size_t n, int bs) {
  return (unsigned)((n + (size_t)bs - 1) / (size_t)bs);
}

extern "C" void kernel_launch(void* const* d_in, const int* in_sizes, int n_in,
                              void* d_out, int out_size, void* d_ws, size_t ws_size,
                              hipStream_t stream) {
  const int* text = (const int*)d_in[0];
  const int* img = (const int*)d_in[1];
  const float* tok_emb = (const float*)d_in[2];
  const float* pos_emb = (const float*)d_in[3];
  const float* qkv_w = (const float*)d_in[4];
  const float* qkv_b = (const float*)d_in[5];
  const float* out_w = (const float*)d_in[6];
  const float* out_b = (const float*)d_in[7];
  const float* ln1_g = (const float*)d_in[8];
  const float* ln1_b = (const float*)d_in[9];
  const float* ln2_g = (const float*)d_in[10];
  const float* ln2_b = (const float*)d_in[11];
  const float* ff1_w = (const float*)d_in[12];
  const float* ff1_b = (const float*)d_in[13];
  const float* ff2_w = (const float*)d_in[14];
  const float* ff2_b = (const float*)d_in[15];
  const float* lnf_g = (const float*)d_in[16];
  const float* lnf_b = (const float*)d_in[17];
  const float* head_w = (const float*)d_in[18];

  char* ws = (char*)d_ws;
  size_t off = 0;
  auto alloc = [&](size_t bytes) -> char* {
    char* p = ws + off;
    off += (bytes + 255) & ~(size_t)255;
    return p;
  };

  const size_t n_tok = (size_t)BB * SS;  // 8704
  _Float16* qkv_w16 = (_Float16*)alloc((size_t)LLAY * DD * 3 * DD * 2);
  _Float16* out_w16 = (_Float16*)alloc((size_t)LLAY * DD * DD * 2);
  _Float16* ff1_w16 = (_Float16*)alloc((size_t)LLAY * DD * FFF * 2);
  _Float16* ff2_w16 = (_Float16*)alloc((size_t)LLAY * FFF * DD * 2);
  _Float16* head_w16 = (_Float16*)alloc((size_t)DD * VII * 2);
  float* x = (float*)alloc(n_tok * DD * 4);
  _Float16* h16 = (_Float16*)alloc(n_tok * DD * 2);
  _Float16* q16 = (_Float16*)alloc(n_tok * DD * 2);
  _Float16* k16 = (_Float16*)alloc(n_tok * DD * 2);
  _Float16* v16 = (_Float16*)alloc(n_tok * DD * 2);
  _Float16* o16 = (_Float16*)alloc(n_tok * DD * 2);
  // big scratch shared by qkv16 / g16 / hsel16 (disjoint lifetimes)
  _Float16* big = (_Float16*)alloc(n_tok * FFF * 2);
  _Float16* qkv16 = big;
  _Float16* g16 = big;
  _Float16* hsel16 = big;

  // --- weight conversions (every call; deterministic)
  {
    size_t n;
    n = (size_t)LLAY * DD * 3 * DD;
    cvt_f32_to_f16_kernel<<<blocks_for(n, 256), 256, 0, stream>>>(qkv_w, qkv_w16, n);
    n = (size_t)LLAY * DD * DD;
    cvt_f32_to_f16_kernel<<<blocks_for(n, 256), 256, 0, stream>>>(out_w, out_w16, n);
    n = (size_t)LLAY * DD * FFF;
    cvt_f32_to_f16_kernel<<<blocks_for(n, 256), 256, 0, stream>>>(ff1_w, ff1_w16, n);
    n = (size_t)LLAY * FFF * DD;
    cvt_f32_to_f16_kernel<<<blocks_for(n, 256), 256, 0, stream>>>(ff2_w, ff2_w16, n);
    n = (size_t)DD * VII;
    cvt_f32_to_f16_kernel<<<blocks_for(n, 256), 256, 0, stream>>>(head_w, head_w16, n);
  }

  // --- embedding
  embed_kernel<<<blocks_for(n_tok * DD, 256), 256, 0, stream>>>(text, img, tok_emb, pos_emb, x);

  const int M = (int)n_tok;  // 8704
  for (int l = 0; l < LLAY; ++l) {
    // LN1 -> h16
    ln_kernel<<<M, 256, 0, stream>>>(x, ln1_g + (size_t)l * DD, ln1_b + (size_t)l * DD, h16);
    // QKV: [M,1024] x [1024,3072] + b -> qkv16
    gemm_kernel<ACT_NONE, true, false, false, true>
        <<<dim3(3 * DD / 128, M / 128), 256, 0, stream>>>(
            h16, qkv_w16 + (size_t)l * DD * 3 * DD, qkv_b + (size_t)l * 3 * DD,
            nullptr, nullptr, qkv16, M, 3 * DD, DD);
    repack_qkv_kernel<<<blocks_for(n_tok * DD, 256), 256, 0, stream>>>(qkv16, q16, k16, v16);
    // attention
    attn_kernel<<<BB * HH * (SS / 16), 32, 16 * SS * sizeof(float), stream>>>(q16, k16, v16, o16);
    // out projection + residual -> x
    gemm_kernel<ACT_NONE, true, true, true, false>
        <<<dim3(DD / 128, M / 128), 256, 0, stream>>>(
            o16, out_w16 + (size_t)l * DD * DD, out_b + (size_t)l * DD,
            x, x, nullptr, M, DD, DD);
    // LN2 -> h16
    ln_kernel<<<M, 256, 0, stream>>>(x, ln2_g + (size_t)l * DD, ln2_b + (size_t)l * DD, h16);
    // FF1 + GELU -> g16
    gemm_kernel<ACT_GELU, true, false, false, true>
        <<<dim3(FFF / 128, M / 128), 256, 0, stream>>>(
            h16, ff1_w16 + (size_t)l * DD * FFF, ff1_b + (size_t)l * FFF,
            nullptr, nullptr, g16, M, FFF, DD);
    // FF2 + residual -> x
    gemm_kernel<ACT_NONE, true, true, true, false>
        <<<dim3(DD / 128, M / 128), 256, 0, stream>>>(
            g16, ff2_w16 + (size_t)l * FFF * DD, ff2_b + (size_t)l * DD,
            x, x, nullptr, M, DD, FFF);
  }

  // final LN -> h16
  ln_kernel<<<M, 256, 0, stream>>>(x, lnf_g, lnf_b, h16);
  // gather output rows (s = T-1 .. S-2) -> hsel16 [8192,1024]
  gather_head_rows_kernel<<<blocks_for((size_t)BB * IIM * DD, 256), 256, 0, stream>>>(h16, hsel16);
  // head GEMM -> d_out (f32), M=8192, N=8192, K=1024, no bias
  gemm_kernel<ACT_NONE, false, false, true, false>
      <<<dim3(VII / 128, (BB * IIM) / 128), 256, 0, stream>>>(
          hsel16, head_w16, nullptr, nullptr, (float*)d_out, nullptr,
          BB * IIM, VII, DD);
}
